// MLP_29420525978344
// MI455X (gfx1250) — matrix operand order repaired and checked by reading, not compiled, not run on hardware
//
#include <hip/hip_runtime.h>
#include <hip/hip_fp16.h>
#include <stdint.h>

typedef __attribute__((ext_vector_type(16))) _Float16 v16h;
typedef __attribute__((ext_vector_type(8)))  _Float16 v8h;
typedef __attribute__((ext_vector_type(8)))  float    v8f;

#define BN_EPS 1e-5f
#define NROWS  262144

// ---------------------------------------------------------------------------
// CDNA5 async global->LDS copy (ASYNCcnt path), per CDNA5 ISA
// ---------------------------------------------------------------------------
__device__ __forceinline__ void async_copy_b128(const void* gsrc, void* ldst) {
    unsigned lds_off = (unsigned)(unsigned long long)(uintptr_t)ldst; // low 32b of flat = LDS offset
    unsigned long long gaddr = (unsigned long long)(uintptr_t)gsrc;
    asm volatile("global_load_async_to_lds_b128 %0, %1, off"
                 :: "v"(lds_off), "v"(gaddr) : "memory");
}
__device__ __forceinline__ void wait_async_zero() {
#if __has_builtin(__builtin_amdgcn_s_wait_asynccnt)
    __builtin_amdgcn_s_wait_asynccnt(0);
#else
    asm volatile("s_wait_asynccnt 0" ::: "memory");
#endif
}

// ---------------------------------------------------------------------------
// Zero the stats accumulator
// ---------------------------------------------------------------------------
__global__ void zero_kernel(float* __restrict__ p, int n) {
    int i = blockIdx.x * blockDim.x + threadIdx.x;
    if (i < n) p[i] = 0.0f;
}

// ---------------------------------------------------------------------------
// Layer 0: a0 = relu(x @ W0 + b0), x: N x 3 (f32), out f16 N x 512.
// Accumulates per-feature sum / sumsq (interleaved) into statsG[2*512].
// Grid: 256 blocks x 256 threads, 1024 rows/block.
// ---------------------------------------------------------------------------
__global__ __launch_bounds__(256) void layer0_kernel(
    const float* __restrict__ x, const float* __restrict__ W0,
    const float* __restrict__ b0, _Float16* __restrict__ Out,
    float* __restrict__ statsG)
{
    __shared__ float sW[3 * 512];
    __shared__ float sb[512];
    __shared__ float sStats[1024];
    const int t = threadIdx.x;
    for (int i = t; i < 1536; i += 256) sW[i] = W0[i];
    for (int i = t; i < 512;  i += 256) sb[i] = b0[i];
    for (int i = t; i < 1024; i += 256) sStats[i] = 0.0f;
    __syncthreads();

    const int chunk = t & 63;   // fixed 8 output columns per thread
    const int sub   = t >> 6;   // 0..3
    const int rowBase = blockIdx.x * 1024;

    float ss[8], sq[8];
#pragma unroll
    for (int j = 0; j < 8; ++j) { ss[j] = 0.0f; sq[j] = 0.0f; }

    for (int i = 0; i < 256; ++i) {
        const int row = rowBase + i * 4 + sub;
        const float* xr = x + (size_t)row * 3;
        const float x0 = xr[0], x1 = xr[1], x2 = xr[2];
        v8h o;
#pragma unroll
        for (int j = 0; j < 8; ++j) {
            const int c = chunk * 8 + j;
            float v = x0 * sW[c] + x1 * sW[512 + c] + x2 * sW[1024 + c] + sb[c];
            v = v > 0.0f ? v : 0.0f;
            o[j] = (_Float16)v;
            ss[j] += v; sq[j] += v * v;
        }
        *(v8h*)(Out + (size_t)row * 512 + chunk * 8) = o;
    }
#pragma unroll
    for (int j = 0; j < 8; ++j) {
        const int c = chunk * 8 + j;
        atomicAdd(&sStats[2 * c],     ss[j]);
        atomicAdd(&sStats[2 * c + 1], sq[j]);
    }
    __syncthreads();
    for (int i = t; i < 1024; i += 256) atomicAdd(&statsG[i], sStats[i]);
}

// ---------------------------------------------------------------------------
// Fold BN(prev layer) into next Linear:
//   s[k] = gamma[k]*rsqrt(var+eps), t[k] = beta[k]-mean*s[k]
//   Wt[c*K+k] = (f16)(s[k]*W[k*M+c])   (transposed, K-contiguous for WMMA B)
//   bp[c]     = b[c] + sum_k t[k]*W[k*M+c]
// Grid: M blocks x 128 threads.
// ---------------------------------------------------------------------------
__global__ __launch_bounds__(128) void fold_kernel(
    const float* __restrict__ stats, const float* __restrict__ gamma,
    const float* __restrict__ beta,  const float* __restrict__ W,
    const float* __restrict__ b,     _Float16* __restrict__ Wt,
    float* __restrict__ bpOut, int K, int M)
{
    const int c = blockIdx.x;
    const float invN = 1.0f / (float)NROWS;
    float part = 0.0f;
    for (int k = threadIdx.x; k < K; k += 128) {
        const float mean = stats[2 * k] * invN;
        const float var  = stats[2 * k + 1] * invN - mean * mean;
        const float s    = gamma[k] * rsqrtf(var + BN_EPS);
        const float tt   = beta[k] - mean * s;
        const float wv   = W[(size_t)k * M + c];
        Wt[(size_t)c * K + k] = (_Float16)(s * wv);
        part += tt * wv;
    }
    __shared__ float red[128];
    red[threadIdx.x] = part;
    __syncthreads();
    for (int s2 = 64; s2 > 0; s2 >>= 1) {
        if (threadIdx.x < s2) red[threadIdx.x] += red[threadIdx.x + s2];
        __syncthreads();
    }
    if (threadIdx.x == 0) bpOut[c] = b[c] + red[0];
}

// ---------------------------------------------------------------------------
// WMMA GEMM + bias + ReLU + BN-stat accumulation.
//   A: N x K (f16, row-major), Wt: M x K (f16, folded, K-contiguous),
//   Out: N x M (f16), statsG: 2*M interleaved (sum,sumsq).
// 256 threads = 8 waves; each wave owns a 16-row strip; 8 row-iters/block
// => 1024 rows/block => 256 blocks.
// Column-group outermost: full-K weight panel staged ONCE per group in LDS
// via async copies (2 barriers per group), then 8 row-iters stream over it.
// A fragments software-pipelined one k-step ahead; k-loop fully unrolled so
// the double-buffer alternates registers with no rotation movs.
// ---------------------------------------------------------------------------
template<int K, int M>
__global__ __launch_bounds__(256) void gemm_bn_relu(
    const _Float16* __restrict__ A, const _Float16* __restrict__ Wt,
    const float* __restrict__ bp, _Float16* __restrict__ Out,
    float* __restrict__ statsG)
{
    constexpr int KP     = K + 8;                      // +16B pad: conflict-free ds reads
    constexpr int CPG    = (K >= 512) ? 32 : ((M >= 64) ? 64 : M); // cols per group
    constexpr int TILES  = CPG / 16;
    constexpr int KSTEPS = K / 32;
    constexpr int ROWIT  = 8;

    __shared__ __align__(16) _Float16 sB[CPG * KP];    // <= ~34 KB
    __shared__ float sStats[2 * M];

    const int t    = threadIdx.x;
    const int w    = t >> 5;
    const int lane = t & 31;
    const int lo   = lane & 15;
    const int hi   = lane >> 4;

    for (int i = t; i < 2 * M; i += 256) sStats[i] = 0.0f;

    const int blockRow0 = blockIdx.x * (ROWIT * 128);

    for (int cg = 0; cg < M; cg += CPG) {
        __syncthreads();  // previous panel fully consumed
        // async-stage full-K weight panel for cols [cg, cg+CPG)
        constexpr int OPS = CPG * K / 8;               // 16-byte transfers
        for (int i = t; i < OPS; i += 256) {
            const int lc = i / (K / 8);
            const int j  = i % (K / 8);
            async_copy_b128(Wt + (size_t)(cg + lc) * K + j * 8,
                            &sB[lc * KP + j * 8]);
        }
        wait_async_zero();
        __syncthreads();

        for (int it = 0; it < ROWIT; ++it) {
            const int rowBase = blockRow0 + it * 128 + w * 16;
            const _Float16* arow = A + (size_t)(rowBase + lo) * K;

            v8f acc[TILES] = {};

            // A fragment (ISA 16-bit 16x32 layout), pipelined one k-step ahead:
            //   halfs 0..7  = K[bk + hi*8 .. +7]
            //   halfs 8..15 = K[bk + 16 + hi*8 .. +7]
            v8h a0 = *(const v8h*)(arow + hi * 8);
            v8h a1 = *(const v8h*)(arow + 16 + hi * 8);
#pragma unroll
            for (int ks = 0; ks < KSTEPS; ++ks) {
                v8h na0, na1;
                if (ks + 1 < KSTEPS) {
                    const int nbk = (ks + 1) * 32;
                    na0 = *(const v8h*)(arow + nbk + hi * 8);
                    na1 = *(const v8h*)(arow + nbk + 16 + hi * 8);
                }
                v16h afr;
#pragma unroll
                for (int e = 0; e < 8; ++e) { afr[e] = a0[e]; afr[e + 8] = a1[e]; }
#pragma unroll
                for (int q = 0; q < TILES; ++q) {
                    // B fragment: col = q*16+lo, K [ks*32 + hi*8 .. +7] and [ks*32 + 16 + hi*8 .. +7]
                    const _Float16* bs = &sB[(q * 16 + lo) * KP + ks * 32 + hi * 8];
                    v8h b0 = *(const v8h*)bs;
                    v8h b1 = *(const v8h*)(bs + 16);
                    v16h bfr;
#pragma unroll
                    for (int e = 0; e < 8; ++e) { bfr[e] = b0[e]; bfr[e + 8] = b1[e]; }
                    acc[q] = __builtin_amdgcn_wmma_f32_16x16x32_f16(
                        false, afr, false, bfr, (short)0, acc[q], false, false);
                }
                a0 = na0; a1 = na1;
            }

            // epilogue: bias + relu + store + stats
#pragma unroll
            for (int q = 0; q < TILES; ++q) {
                const int col = cg + q * 16 + lo;
                const float bv = bp[col];
                float s = 0.0f, s2 = 0.0f;
#pragma unroll
                for (int r = 0; r < 8; ++r) {
                    float v = acc[q][r] + bv;
                    v = v > 0.0f ? v : 0.0f;
                    const int row = rowBase + hi * 8 + r;  // C/D layout: M = r + hi*8
                    Out[(size_t)row * M + col] = (_Float16)v;
                    s += v; s2 += v * v;
                }
                atomicAdd(&sStats[2 * col],     s);
                atomicAdd(&sStats[2 * col + 1], s2);
            }
        }
    }
    __syncthreads();
    for (int i = t; i < 2 * M; i += 256) atomicAdd(&statsG[i], sStats[i]);
}

// ---------------------------------------------------------------------------
// Tail layers (K<=16): VALU dot products; optional final f32 output.
// 64 blocks x 256 threads x 16 rows.
// ---------------------------------------------------------------------------
template<int K, int M, bool FINAL>
__global__ __launch_bounds__(256) void tail_layer(
    const _Float16* __restrict__ A, const _Float16* __restrict__ Wt,
    const float* __restrict__ bp, _Float16* __restrict__ Out,
    float* __restrict__ OutF, float* __restrict__ statsG)
{
    __shared__ _Float16 sW[M * K];
    __shared__ float sb[M];
    __shared__ float sStats[2 * M];
    const int t = threadIdx.x;
    for (int i = t; i < M * K; i += 256) sW[i] = Wt[i];
    for (int i = t; i < M;     i += 256) sb[i] = bp[i];
    for (int i = t; i < 2 * M; i += 256) sStats[i] = 0.0f;
    __syncthreads();

    float ss[M], sq[M];
#pragma unroll
    for (int c = 0; c < M; ++c) { ss[c] = 0.0f; sq[c] = 0.0f; }

    const int base = blockIdx.x * 4096;
    for (int i = 0; i < 16; ++i) {
        const int row = base + i * 256 + t;
        const _Float16* ar = A + (size_t)row * K;
        _Float16 a[K];
#pragma unroll
        for (int k = 0; k < K; ++k) a[k] = ar[k];
#pragma unroll
        for (int c = 0; c < M; ++c) {
            float acc = sb[c];
#pragma unroll
            for (int k = 0; k < K; ++k) acc += (float)a[k] * (float)sW[c * K + k];
            if (FINAL) {
                OutF[row] = acc;
            } else {
                acc = acc > 0.0f ? acc : 0.0f;
                Out[(size_t)row * M + c] = (_Float16)acc;
                ss[c] += acc; sq[c] += acc * acc;
            }
        }
    }
    if (!FINAL) {
#pragma unroll
        for (int c = 0; c < M; ++c) {
            atomicAdd(&sStats[2 * c],     ss[c]);
            atomicAdd(&sStats[2 * c + 1], sq[c]);
        }
        __syncthreads();
        for (int i = t; i < 2 * M; i += 256) atomicAdd(&statsG[i], sStats[i]);
    }
}

// ---------------------------------------------------------------------------
extern "C" void kernel_launch(void* const* d_in, const int* in_sizes, int n_in,
                              void* d_out, int out_size, void* d_ws, size_t ws_size,
                              hipStream_t stream) {
    (void)in_sizes; (void)n_in; (void)out_size; (void)ws_size;
    const float* x = (const float*)d_in[0];
    const float *W[11], *B[11], *G[10], *Be[10];
    for (int i = 0; i < 11; ++i) { W[i] = (const float*)d_in[1 + i]; B[i] = (const float*)d_in[12 + i]; }
    for (int i = 0; i < 10; ++i) { G[i] = (const float*)d_in[23 + i]; Be[i] = (const float*)d_in[33 + i]; }

    char* ws = (char*)d_ws;
    _Float16* aA = (_Float16*)ws;  ws += (size_t)NROWS * 512 * 2;   // ping (<=512 wide)
    _Float16* aB = (_Float16*)ws;  ws += (size_t)NROWS * 256 * 2;   // pong (<=256 wide)
    _Float16* wt = (_Float16*)ws;  ws += (size_t)512 * 256 * 2;     // folded weights (max)
    float* bp    = (float*)ws;     ws += 512 * 4;                    // folded bias
    float* stats = (float*)ws;     ws += 1024 * 4;                   // interleaved sum/sumsq
    float* outF  = (float*)d_out;

#define ZERO_STATS() zero_kernel<<<4, 256, 0, stream>>>(stats, 1024)

    // L0: 3 -> 512 (f32 VALU, memory-bound) + stats of a0
    ZERO_STATS();
    layer0_kernel<<<256, 256, 0, stream>>>(x, W[0], B[0], aA, stats);

    // L1: 512 -> 256 (WMMA)
    fold_kernel<<<256, 128, 0, stream>>>(stats, G[0], Be[0], W[1], B[1], wt, bp, 512, 256);
    ZERO_STATS();
    gemm_bn_relu<512, 256><<<256, 256, 0, stream>>>(aA, wt, bp, aB, stats);

    // L2: 256 -> 128
    fold_kernel<<<128, 128, 0, stream>>>(stats, G[1], Be[1], W[2], B[2], wt, bp, 256, 128);
    ZERO_STATS();
    gemm_bn_relu<256, 128><<<256, 256, 0, stream>>>(aB, wt, bp, aA, stats);

    // L3: 128 -> 64
    fold_kernel<<<64, 128, 0, stream>>>(stats, G[2], Be[2], W[3], B[3], wt, bp, 128, 64);
    ZERO_STATS();
    gemm_bn_relu<128, 64><<<256, 256, 0, stream>>>(aA, wt, bp, aB, stats);

    // L4: 64 -> 32
    fold_kernel<<<32, 128, 0, stream>>>(stats, G[3], Be[3], W[4], B[4], wt, bp, 64, 32);
    ZERO_STATS();
    gemm_bn_relu<64, 32><<<256, 256, 0, stream>>>(aB, wt, bp, aA, stats);

    // L5: 32 -> 16
    fold_kernel<<<16, 128, 0, stream>>>(stats, G[4], Be[4], W[5], B[5], wt, bp, 32, 16);
    ZERO_STATS();
    gemm_bn_relu<32, 16><<<256, 256, 0, stream>>>(aA, wt, bp, aB, stats);

    // L6: 16 -> 8 (tail)
    fold_kernel<<<8, 128, 0, stream>>>(stats, G[5], Be[5], W[6], B[6], wt, bp, 16, 8);
    ZERO_STATS();
    tail_layer<16, 8, false><<<64, 256, 0, stream>>>(aB, wt, bp, aA, nullptr, stats);

    // L7: 8 -> 4
    fold_kernel<<<4, 128, 0, stream>>>(stats, G[6], Be[6], W[7], B[7], wt, bp, 8, 4);
    ZERO_STATS();
    tail_layer<8, 4, false><<<64, 256, 0, stream>>>(aA, wt, bp, aB, nullptr, stats);

    // L8: 4 -> 4
    fold_kernel<<<4, 128, 0, stream>>>(stats, G[7], Be[7], W[8], B[8], wt, bp, 4, 4);
    ZERO_STATS();
    tail_layer<4, 4, false><<<64, 256, 0, stream>>>(aB, wt, bp, aA, nullptr, stats);

    // L9: 4 -> 4
    fold_kernel<<<4, 128, 0, stream>>>(stats, G[8], Be[8], W[9], B[9], wt, bp, 4, 4);
    ZERO_STATS();
    tail_layer<4, 4, false><<<64, 256, 0, stream>>>(aA, wt, bp, aB, nullptr, stats);

    // L10: final 4 -> 1 (BN9 folded in; no relu/stats), f32 output
    fold_kernel<<<1, 128, 0, stream>>>(stats, G[9], Be[9], W[10], B[10], wt, bp, 4, 1);
    tail_layer<4, 1, true><<<64, 256, 0, stream>>>(aB, wt, bp, nullptr, outF, stats);

#undef ZERO_STATS
}
